// TemporalModule_13494787244852
// MI455X (gfx1250) — matrix operand order repaired and checked by reading, not compiled, not run on hardware
//
#include <hip/hip_runtime.h>

// ---------------------------------------------------------------------------
// Fused temporal-attention kernel for MI455X (gfx1250, wave32, WMMA bf16).
//   theta = Xs@Wa^T+ba ; phi = Xs@Wb^T+bb ; attn = theta@phi^T ;
//   out   = attn @ (Xs@Wg^T + bg)
// One wave per object s; 8 waves (8 objects) per workgroup share bf16
// weights staged in LDS. All matrix math via v_wmma_f32_16x16x32_bf16.
// ---------------------------------------------------------------------------

typedef __attribute__((ext_vector_type(16))) __bf16 v16bf;
typedef __attribute__((ext_vector_type(8)))  float  v8f;

#define T_DIM 32
#define S_DIM 4096
#define F_DIM 256
#define A_DIM 64

#define WAVES    8            // 8 objects per workgroup
#define WLD      264          // padded LDS row (bf16) for weights (132-dword stride)
#define TLD      66           // padded row for theta/phi scratch
#define ALD      34           // padded row for attn scratch
#define FTLD     34           // padded row for feats-tile scratch (stored [f][u])

// LDS layout in bf16 elements
#define OFF_WA   0
#define OFF_WB   (OFF_WA + 64 * WLD)
#define OFF_WG   (OFF_WB + 64 * WLD)
#define OFF_TH   (OFF_WG + 256 * WLD)
#define OFF_PH   (OFF_TH + WAVES * 32 * TLD)
#define OFF_AT   (OFF_PH + WAVES * 32 * TLD)
#define OFF_FT   (OFF_AT + WAVES * 32 * ALD)
#define LDS_TOT  (OFF_FT + WAVES * 16 * FTLD)   // ~290 KB of 320 KB/WGP

// ---- WMMA fragment builders -----------------------------------------------
// A-operand (16x32 bf16): lane L (0..15|16..31) holds row m0+(L&15);
// element e -> K = k0 + (e/8)*16 + (L>>4)*8 + (e%8)   (per CDNA5 ISA 7.12.2)
__device__ __forceinline__ v16bf load_a_global_f32(const float* __restrict__ base,
                                                   long ld, int m0, int k0, int lane) {
  const int r = lane & 15, h = lane >> 4;
  const float* p = base + (long)(m0 + r) * ld + k0 + h * 8;
  v16bf a;
#pragma unroll
  for (int j = 0; j < 8; ++j) a[j] = (__bf16)p[j];
#pragma unroll
  for (int j = 0; j < 8; ++j) a[8 + j] = (__bf16)p[16 + j];
  return a;
}

__device__ __forceinline__ v16bf load_a_lds(const __bf16* base, int ld,
                                            int m0, int k0, int lane) {
  const int r = lane & 15, h = lane >> 4;
  const __bf16* p = base + (m0 + r) * ld + k0 + h * 8;
  v16bf a;
#pragma unroll
  for (int j = 0; j < 8; ++j) a[j] = p[j];
#pragma unroll
  for (int j = 0; j < 8; ++j) a[8 + j] = p[16 + j];
  return a;
}

// B-operand (32x16 bf16) built from a row-major [n][k] LDS matrix, i.e.
// B[k][n] = W[n0+n][k0+k]. Lane L: N = L&15, element e -> K = (e/8)*16 + (L>>4)*8 + (e%8).
// Per lane this is two runs of 8 contiguous bf16 (16 B each).
__device__ __forceinline__ v16bf load_b_ldsT(const __bf16* w, int ld,
                                             int n0, int k0, int lane) {
  const int n = lane & 15, h = lane >> 4;
  const __bf16* p = w + (n0 + n) * ld + k0 + h * 8;
  v16bf b;
#pragma unroll
  for (int e = 0; e < 8; ++e) { b[e] = p[e]; b[8 + e] = p[16 + e]; }
  return b;
}

__device__ __forceinline__ v8f wmma_bf16(v16bf a, v16bf b, v8f c) {
  return __builtin_amdgcn_wmma_f32_16x16x32_bf16(
      /*neg_a=*/false, a, /*neg_b=*/false, b,
      /*c_mod=*/(short)0, c, /*reuse_a=*/false, /*reuse_b=*/false);
}

// ---------------------------------------------------------------------------
__global__ __launch_bounds__(32 * WAVES, 1)
void temporal_fused_kernel(const float* __restrict__ x,   // [T*S, F]
                           const float* __restrict__ Wa,  // [A, F]
                           const float* __restrict__ ba,  // [A]
                           const float* __restrict__ Wb,  // [A, F]
                           const float* __restrict__ bb,  // [A]
                           const float* __restrict__ Wg,  // [F, F]
                           const float* __restrict__ bg,  // [F]
                           float* __restrict__ out) {     // [T*S, F]
  __shared__ __bf16 lds[LDS_TOT];

  const int tid  = threadIdx.x;
  const int lane = tid & 31;
  const int wave = tid >> 5;
  const int s    = blockIdx.x * WAVES + wave;

  // ---- cooperative weight preload: fp32 global -> bf16 LDS (padded) ----
  for (int i = tid; i < 64 * F_DIM; i += 32 * WAVES) {
    const int n = i >> 8, k = i & (F_DIM - 1);
    lds[OFF_WA + n * WLD + k] = (__bf16)Wa[i];
    lds[OFF_WB + n * WLD + k] = (__bf16)Wb[i];
  }
  for (int i = tid; i < 256 * F_DIM; i += 32 * WAVES) {
    const int n = i >> 8, k = i & (F_DIM - 1);
    lds[OFF_WG + n * WLD + k] = (__bf16)Wg[i];
  }
  __syncthreads();

  __bf16* sTh = lds + OFF_TH + wave * 32 * TLD;   // theta_s [t][a]
  __bf16* sPh = lds + OFF_PH + wave * 32 * TLD;   // phi_s   [u][a]
  __bf16* sAt = lds + OFF_AT + wave * 32 * ALD;   // attn_s  [t][u]
  __bf16* sFt = lds + OFF_FT + wave * 16 * FTLD;  // feats tile, transposed [f][u]

  const float* xs  = x + (long)s * F_DIM;         // row m (=t or u) stride S*F
  const long   ldx = (long)S_DIM * F_DIM;
  const int    col16 = lane & 15, half = lane >> 4;

  // ---- Phase 1: theta / phi projections (K = 256) ----
#pragma unroll
  for (int mt = 0; mt < 2; ++mt) {
#pragma unroll
    for (int nt = 0; nt < A_DIM / 16; ++nt) {
      v8f ct = {}, cp = {};
#pragma unroll
      for (int kt = 0; kt < F_DIM / 32; ++kt) {
        v16bf a  = load_a_global_f32(xs, ldx, mt * 16, kt * 32, lane);
        v16bf bA = load_b_ldsT(lds + OFF_WA, WLD, nt * 16, kt * 32, lane);
        v16bf bB = load_b_ldsT(lds + OFF_WB, WLD, nt * 16, kt * 32, lane);
        ct = wmma_bf16(a, bA, ct);
        cp = wmma_bf16(a, bB, cp);
      }
      const int   col = nt * 16 + col16;
      const float bia = ba[col], bib = bb[col];
#pragma unroll
      for (int r = 0; r < 8; ++r) {
        const int row = mt * 16 + r + 8 * half;   // C layout: M = r + 8*(lane/16)
        sTh[row * TLD + col] = (__bf16)(ct[r] + bia);
        sPh[row * TLD + col] = (__bf16)(cp[r] + bib);
      }
    }
  }
  // wave-private scratch: no workgroup barrier needed, only dscnt ordering

  // ---- Phase 2: attn = theta @ phi^T  (32x32, K = 64) ----
#pragma unroll
  for (int mt = 0; mt < 2; ++mt) {
#pragma unroll
    for (int nt = 0; nt < 2; ++nt) {
      v8f c = {};
#pragma unroll
      for (int kt = 0; kt < A_DIM / 32; ++kt) {
        v16bf a = load_a_lds(sTh, TLD, mt * 16, kt * 32, lane);
        v16bf b = load_b_ldsT(sPh, TLD, nt * 16, kt * 32, lane);
        c = wmma_bf16(a, b, c);
      }
      const int col = nt * 16 + col16;
#pragma unroll
      for (int r = 0; r < 8; ++r)
        sAt[(mt * 16 + r + 8 * half) * ALD + col] = (__bf16)c[r];
    }
  }

  // ---- Phase 3: per 16-wide f-tile: feats tile, then out = attn @ feats ----
  for (int ft = 0; ft < F_DIM / 16; ++ft) {
    // feats[u, ftile] = Xs @ Wg^T + bg ; store transposed [f][u] for B reload
    const float bi = bg[ft * 16 + col16];
#pragma unroll
    for (int mt = 0; mt < 2; ++mt) {
      v8f c = {};
#pragma unroll
      for (int kt = 0; kt < F_DIM / 32; ++kt) {
        v16bf a = load_a_global_f32(xs, ldx, mt * 16, kt * 32, lane);  // L2 hit
        v16bf b = load_b_ldsT(lds + OFF_WG, WLD, ft * 16, kt * 32, lane);
        c = wmma_bf16(a, b, c);
      }
#pragma unroll
      for (int r = 0; r < 8; ++r) {
        const int u = mt * 16 + r + 8 * half;
        sFt[col16 * FTLD + u] = (__bf16)(c[r] + bi);
      }
    }
    // out tile: single K=32 step, A = attn (LDS), B = feats tile (LDS, [f][u])
#pragma unroll
    for (int mt = 0; mt < 2; ++mt) {
      v16bf a = load_a_lds(sAt, ALD, mt * 16, 0, lane);
      v16bf b = load_b_ldsT(sFt, FTLD, 0, 0, lane);
      v8f c = {};
      c = wmma_bf16(a, b, c);
#pragma unroll
      for (int r = 0; r < 8; ++r) {
        const int t = mt * 16 + r + 8 * half;
        __builtin_nontemporal_store(
            c[r], &out[((long)t * S_DIM + s) * F_DIM + ft * 16 + col16]);
      }
    }
  }
}

// ---------------------------------------------------------------------------
extern "C" void kernel_launch(void* const* d_in, const int* in_sizes, int n_in,
                              void* d_out, int out_size, void* d_ws, size_t ws_size,
                              hipStream_t stream) {
  // setup_inputs order:
  // 0 batch_data [T*S,F], 1 xywh (dead), 2 Wa, 3 ba, 4 Wb, 5 bb,
  // 6 Wg, 7 bg, 8 Wh (dead), 9 bh (dead)
  const float* x  = (const float*)d_in[0];
  const float* Wa = (const float*)d_in[2];
  const float* ba = (const float*)d_in[3];
  const float* Wb = (const float*)d_in[4];
  const float* bb = (const float*)d_in[5];
  const float* Wg = (const float*)d_in[6];
  const float* bg = (const float*)d_in[7];
  float* out = (float*)d_out;

  dim3 grid(S_DIM / WAVES);      // 512 workgroups, one 8-object tile each
  dim3 block(32 * WAVES);        // 8 waves of 32
  temporal_fused_kernel<<<grid, block, 0, stream>>>(x, Wa, ba, Wb, bb, Wg, bg, out);
}
